// MoEFFN_25640954757690
// MI455X (gfx1250) — compile-verified, matrix operand order checked by reading, not run on hardware
//
#include <hip/hip_runtime.h>
#include <hip/hip_bf16.h>
#include <math.h>

// ---------------- static problem config ----------------
#define N_TOK 2048      // B*T
#define DIM   768
#define SHID  3072
#define NEXP  16
#define HID   384
#define TOPK  4
#define NNULL 16
#define RHO   0.5f

typedef __attribute__((ext_vector_type(16))) __bf16 v16bf;
typedef __attribute__((ext_vector_type(8)))  __bf16 v8bf;
typedef __attribute__((ext_vector_type(8)))  float  v8f;
typedef __attribute__((ext_vector_type(4)))  unsigned int u32x4;
typedef __attribute__((ext_vector_type(4)))  int i32x4;
typedef __attribute__((ext_vector_type(8)))  int i32x8;

// TDM availability (device pass only)
#if defined(__gfx1250__) && __has_builtin(__builtin_amdgcn_tensor_load_to_lds)
#define HAVE_TDM 1
#else
#define HAVE_TDM 0
#endif

// round-to-nearest-even f32 -> bf16 (one-time conversion passes / epilogues only)
__device__ __forceinline__ __bf16 f2bf(float f) {
  unsigned u = __builtin_bit_cast(unsigned, f);
  unsigned r = u + 0x7FFFu + ((u >> 16) & 1u);
  unsigned short h = (unsigned short)(r >> 16);
  return __builtin_bit_cast(__bf16, h);
}

// ---- WMMA bf16 fragment loader: element (m,k) at p[m*ld + k], k-contiguous ----
// lane l: m=l&15, half=l>>4 ; 16 halves: k = half*8+0..7 and k = 16+half*8+0..7
__device__ __forceinline__ v16bf load_frag_bf16(const __bf16* __restrict__ p, int ld) {
  int lane = threadIdx.x & 31;
  int m = lane & 15, half = lane >> 4;
  const __bf16* q = p + m * ld + half * 8;
  v8bf lo = *(const v8bf*)q;
  v8bf hi = *(const v8bf*)(q + 16);
  v16bf r;
#pragma unroll
  for (int i = 0; i < 8; ++i) { r[i] = lo[i]; r[i + 8] = hi[i]; }
  return r;
}

#define WMMA_BF16(a, b, c) \
  __builtin_amdgcn_wmma_f32_16x16x32_bf16(false, (a), false, (b), (short)0, (c), false, false)

__device__ __forceinline__ float silu(float g) { return g / (1.f + __expf(-g)); }

#if HAVE_TDM
__device__ __forceinline__ void tdm_wait0() {
#if __has_builtin(__builtin_amdgcn_s_wait_tensorcnt)
  __builtin_amdgcn_s_wait_tensorcnt(0);
#else
  asm volatile("s_wait_tensorcnt 0x0" ::: "memory");
#endif
}

// 2-D bf16 tile load: tile_rows x tile_cols elements from global (row stride
// row_stride elems) into contiguous LDS at byte offset lds_off.  D# per ISA ch.8.
__device__ __forceinline__ void tdm_load_2d_bf16(const __bf16* gsrc, unsigned lds_off,
                                                 int tile_cols, int tile_rows,
                                                 int tensor_cols, int tensor_rows,
                                                 int row_stride) {
  unsigned long long ga = (unsigned long long)gsrc;
  u32x4 g0;
  g0[0] = 1u;                                                   // count=1, user D#
  g0[1] = lds_off;                                              // lds_addr (bytes)
  g0[2] = (unsigned)ga;                                         // global_addr[31:0]
  g0[3] = (unsigned)((ga >> 32) & 0x01FFFFFFull) | (2u << 30);  // addr[56:32] | type=2
  i32x8 g1 = {};
  g1[0] = 1 << 16;                                              // data_size=1 (2 bytes)
  g1[1] = (tensor_cols & 0xFFFF) << 16;                         // tensor_dim0[15:0]
  g1[2] = ((tensor_cols >> 16) & 0xFFFF) | ((tensor_rows & 0xFFFF) << 16);
  g1[3] = ((tensor_rows >> 16) & 0xFFFF) | ((tile_cols & 0xFFFF) << 16); // tile_dim0
  g1[4] = tile_rows & 0xFFFF;                                   // tile_dim1 (tile_dim2=0)
  g1[5] = row_stride;                                           // tensor_dim0_stride[31:0]
  g1[6] = 0; g1[7] = 0;
#if __has_include(<hip/amd_detail/amd_gfx1250_TDM.h>)
  i32x8 z8 = {};
  i32x4 z4 = {};
  __builtin_amdgcn_tensor_load_to_lds(g0, g1, z4, z4, z8, 0);   // 6-arg toolchain
#else
  i32x4 z4 = {};
  __builtin_amdgcn_tensor_load_to_lds(g0, g1, z4, z4, 0);       // 5-arg (ROCm 7.2)
#endif
}
#endif  // HAVE_TDM

// ---------------- conversion pre-passes ----------------
__global__ void cvt_bf16_v4(const float* __restrict__ src, __bf16* __restrict__ dst, int n4) {
  int i = blockIdx.x * blockDim.x + threadIdx.x;
  if (i >= n4) return;
  const float4 v = ((const float4*)src)[i];
  __bf16* d = dst + 4 * (size_t)i;
  d[0] = f2bf(v.x); d[1] = f2bf(v.y); d[2] = f2bf(v.z); d[3] = f2bf(v.w);
}

// batched transpose+convert: src [Z][R][C] f32 -> dst [Z][C][R] bf16, via LDS tile
__global__ void transpose_cvt(const float* __restrict__ src, __bf16* __restrict__ dst,
                              int R, int C) {
  __shared__ __bf16 tile[16][17];
  const float* s = src + (size_t)blockIdx.z * R * C;
  __bf16*      d = dst + (size_t)blockIdx.z * R * C;
  int c0 = blockIdx.x * 16, r0 = blockIdx.y * 16;
  int tx = threadIdx.x, ty = threadIdx.y;
  tile[ty][tx] = f2bf(s[(size_t)(r0 + ty) * C + c0 + tx]);
  __syncthreads();
  d[(size_t)(c0 + ty) * R + r0 + tx] = tile[tx][ty];
}

// ---------------- zero aux accumulators ----------------
__global__ void init_aux(float* __restrict__ aux) {
  if (threadIdx.x < 64) aux[threadIdx.x] = 0.f;
}

// ---------------- router ----------------
// aux: [0..15] sum softmax(real) ; [16..31] top-k counts ; [32] null slots ; [33] sum lse^2
__global__ void moe_router(const float* __restrict__ x,
                           const float* __restrict__ gate_w,
                           const float* __restrict__ logit_bias,
                           const float* __restrict__ null_logit_p,
                           float* __restrict__ combine,
                           float* __restrict__ aux) {
  int wave = threadIdx.x >> 5;
  int lane = threadIdx.x & 31;
  int t = blockIdx.x * (blockDim.x >> 5) + wave;
  if (t >= N_TOK) return;

  float acc[NEXP];
#pragma unroll
  for (int e = 0; e < NEXP; ++e) acc[e] = 0.f;
  const float* xr = x + t * DIM;
  for (int k0 = 0; k0 < DIM; k0 += 32) {
    float xv = xr[k0 + lane];
#pragma unroll
    for (int e = 0; e < NEXP; ++e)
      acc[e] += xv * gate_w[e * DIM + k0 + lane];
  }
#pragma unroll
  for (int e = 0; e < NEXP; ++e) {
#pragma unroll
    for (int off = 16; off > 0; off >>= 1)
      acc[e] += __shfl_xor(acc[e], off, 32);
  }

  if (lane == 0) {
    float nl = null_logit_p[0];
    float lg[NEXP];
    float mr = -1e30f;
#pragma unroll
    for (int e = 0; e < NEXP; ++e) { lg[e] = acc[e] + logit_bias[e]; mr = fmaxf(mr, lg[e]); }
    float mall = fmaxf(mr, nl);
    float se = 0.f;
#pragma unroll
    for (int e = 0; e < NEXP; ++e) se += __expf(lg[e] - mall);
    float denom = se + (float)NNULL * __expf(nl - mall);

    float dr = 0.f;
#pragma unroll
    for (int e = 0; e < NEXP; ++e) dr += __expf(lg[e] - mr);
#pragma unroll
    for (int e = 0; e < NEXP; ++e) atomicAdd(&aux[e], __expf(lg[e] - mr) / dr);

    float lz = mall + __logf(denom);
    atomicAdd(&aux[33], lz * lz);

    bool used[NEXP];
#pragma unroll
    for (int e = 0; e < NEXP; ++e) used[e] = false;
    int   sel[TOPK];
    float slog[TOPK];
#pragma unroll
    for (int j = 0; j < TOPK; ++j) {
      float best = -1e30f; int bi = 0;
      for (int e = 0; e < NEXP; ++e)
        if (!used[e] && lg[e] > best) { best = lg[e]; bi = e; }
      used[bi] = true; sel[j] = bi; slog[j] = best;
    }
    float wv[TOPK]; float wsum = 0.f; int nreal = 0;
#pragma unroll
    for (int j = 0; j < TOPK; ++j) {
      if (slog[j] >= nl) { wv[nreal] = __expf(slog[j] - mall) / denom; wsum += wv[nreal]; ++nreal; }
    }
    float cw[NEXP];
#pragma unroll
    for (int e = 0; e < NEXP; ++e) cw[e] = 0.f;
    float inv = 1.f / fmaxf(wsum, 1e-6f);
    for (int j = 0; j < nreal; ++j) { cw[sel[j]] = wv[j] * inv; atomicAdd(&aux[16 + sel[j]], 1.f); }
    if (nreal < TOPK) atomicAdd(&aux[32], (float)(TOPK - nreal));
#pragma unroll
    for (int e = 0; e < NEXP; ++e) combine[t * NEXP + e] = cw[e];
  }
}

// ---------------- shared up/gate -> Hs (bf16), 32x32 tile per wave ----------------
__global__ void shared_mlp1(const __bf16* __restrict__ xb,   // [N, D]
                            const __bf16* __restrict__ wg,   // [SH, D]
                            const __bf16* __restrict__ wu,   // [SH, D]
                            __bf16* __restrict__ Hs) {       // [N, SH]
  int wid = blockIdx.x * (blockDim.x >> 5) + (threadIdx.x >> 5);
  const int cB = SHID / 32;
  int rb = wid / cB, cb = wid % cB;
  if (rb >= N_TOK / 32) return;
  int row0 = rb * 32, col0 = cb * 32;

  v8f cg00 = {}, cg01 = {}, cg10 = {}, cg11 = {};
  v8f cu00 = {}, cu01 = {}, cu10 = {}, cu11 = {};
  const __bf16* a0 = xb + (size_t)row0 * DIM;
  const __bf16* a1 = a0 + 16 * DIM;
  const __bf16* g0 = wg + (size_t)col0 * DIM;
  const __bf16* g1 = g0 + 16 * DIM;
  const __bf16* u0 = wu + (size_t)col0 * DIM;
  const __bf16* u1 = u0 + 16 * DIM;
  for (int k0 = 0; k0 < DIM; k0 += 32) {
    __builtin_prefetch(a0 + k0 + 64, 0, 3);
    v16bf A0 = load_frag_bf16(a0 + k0, DIM);
    v16bf A1 = load_frag_bf16(a1 + k0, DIM);
    v16bf G0 = load_frag_bf16(g0 + k0, DIM);
    v16bf G1 = load_frag_bf16(g1 + k0, DIM);
    v16bf U0 = load_frag_bf16(u0 + k0, DIM);
    v16bf U1 = load_frag_bf16(u1 + k0, DIM);
    cg00 = WMMA_BF16(A0, G0, cg00); cg01 = WMMA_BF16(A0, G1, cg01);
    cg10 = WMMA_BF16(A1, G0, cg10); cg11 = WMMA_BF16(A1, G1, cg11);
    cu00 = WMMA_BF16(A0, U0, cu00); cu01 = WMMA_BF16(A0, U1, cu01);
    cu10 = WMMA_BF16(A1, U0, cu10); cu11 = WMMA_BF16(A1, U1, cu11);
  }
  int lane = threadIdx.x & 31, n = lane & 15, half = lane >> 4;
  v8f* G[4] = { &cg00, &cg01, &cg10, &cg11 };
  v8f* U[4] = { &cu00, &cu01, &cu10, &cu11 };
#pragma unroll
  for (int q = 0; q < 4; ++q) {
    int r = (q >> 1) * 16, c = (q & 1) * 16;
#pragma unroll
    for (int v = 0; v < 8; ++v) {
      int m = row0 + r + half * 8 + v;
      Hs[(size_t)m * SHID + col0 + c + n] = f2bf(silu((*G[q])[v]) * (*U[q])[v]);
    }
  }
}

// ---------------- routed up/gate (all experts), scale by combine -> Hr ----------------
__global__ void routed_mlp1(const __bf16* __restrict__ xb,     // [N, D]
                            const __bf16* __restrict__ wgt,    // [E, H, D]
                            const __bf16* __restrict__ wut,    // [E, H, D]
                            const float*  __restrict__ combine,// [N, E]
                            __bf16* __restrict__ Hr) {         // [E, N, H]
  int wid = blockIdx.x * (blockDim.x >> 5) + (threadIdx.x >> 5);
  const int rB = N_TOK / 32, cB = HID / 32;
  int e = wid / (rB * cB);
  if (e >= NEXP) return;
  int rem = wid % (rB * cB);
  int rb = rem / cB, cb = rem % cB;
  int row0 = rb * 32, col0 = cb * 32;

  v8f cg00 = {}, cg01 = {}, cg10 = {}, cg11 = {};
  v8f cu00 = {}, cu01 = {}, cu10 = {}, cu11 = {};
  const __bf16* a0 = xb + (size_t)row0 * DIM;
  const __bf16* a1 = a0 + 16 * DIM;
  const __bf16* g0 = wgt + ((size_t)e * HID + col0) * DIM;
  const __bf16* g1 = g0 + 16 * DIM;
  const __bf16* u0 = wut + ((size_t)e * HID + col0) * DIM;
  const __bf16* u1 = u0 + 16 * DIM;
  for (int k0 = 0; k0 < DIM; k0 += 32) {
    __builtin_prefetch(g0 + k0 + 64, 0, 3);
    v16bf A0 = load_frag_bf16(a0 + k0, DIM);
    v16bf A1 = load_frag_bf16(a1 + k0, DIM);
    v16bf G0 = load_frag_bf16(g0 + k0, DIM);
    v16bf G1 = load_frag_bf16(g1 + k0, DIM);
    v16bf U0 = load_frag_bf16(u0 + k0, DIM);
    v16bf U1 = load_frag_bf16(u1 + k0, DIM);
    cg00 = WMMA_BF16(A0, G0, cg00); cg01 = WMMA_BF16(A0, G1, cg01);
    cg10 = WMMA_BF16(A1, G0, cg10); cg11 = WMMA_BF16(A1, G1, cg11);
    cu00 = WMMA_BF16(A0, U0, cu00); cu01 = WMMA_BF16(A0, U1, cu01);
    cu10 = WMMA_BF16(A1, U0, cu10); cu11 = WMMA_BF16(A1, U1, cu11);
  }
  int lane = threadIdx.x & 31, n = lane & 15, half = lane >> 4;
  v8f* G[4] = { &cg00, &cg01, &cg10, &cg11 };
  v8f* U[4] = { &cu00, &cu01, &cu10, &cu11 };
#pragma unroll
  for (int q = 0; q < 4; ++q) {
    int r = (q >> 1) * 16, c = (q & 1) * 16;
#pragma unroll
    for (int v = 0; v < 8; ++v) {
      int m = row0 + r + half * 8 + v;
      float cmb = combine[m * NEXP + e];
      Hr[((size_t)e * N_TOK + m) * HID + col0 + c + n] = f2bf(cmb * silu((*G[q])[v]) * (*U[q])[v]);
    }
  }
}

// ---------------- shared down -> writes y (TDM-staged A operand) ----------------
// grid: (N/32, 3) ; block: 256 = 8 waves ; wave w computes cols (blockIdx.y*8+w)*32
#define KCH 64   // K elements staged per chunk (32 rows x 64 cols bf16 = 4 KB)
__global__ void shared_down(const __bf16* __restrict__ Hs,   // [N, SH]
                            const __bf16* __restrict__ wd,   // [D, SH]
                            float* __restrict__ out) {       // [N, D]
  __shared__ __align__(16) __bf16 abuf[2][32 * KCH];
  int wave = threadIdx.x >> 5;
  int rb = blockIdx.x;
  int cb = blockIdx.y * 8 + wave;
  int row0 = rb * 32, col0 = cb * 32;

  const __bf16* asrc = Hs + (size_t)row0 * SHID;
  const __bf16* b0 = wd + (size_t)col0 * SHID;
  const __bf16* b1 = b0 + 16 * SHID;
  v8f c00 = {}, c01 = {}, c10 = {}, c11 = {};

#if HAVE_TDM
  const int NCH = SHID / KCH;                     // 48 chunks
  if (wave == 0) {
    tdm_load_2d_bf16(asrc, (unsigned)(unsigned long long)&abuf[0][0],
                     KCH, 32, SHID, N_TOK, SHID);
    tdm_wait0();
  }
  __syncthreads();
  for (int ch = 0; ch < NCH; ++ch) {
    int k0 = ch * KCH;
    if (wave == 0 && ch + 1 < NCH)
      tdm_load_2d_bf16(asrc + k0 + KCH, (unsigned)(unsigned long long)&abuf[(ch + 1) & 1][0],
                       KCH, 32, SHID, N_TOK, SHID);
    const __bf16* at = &abuf[ch & 1][0];
#pragma unroll
    for (int kk = 0; kk < KCH; kk += 32) {
      v16bf A0 = load_frag_bf16(at + kk, KCH);            // rows 0..15 of tile
      v16bf A1 = load_frag_bf16(at + 16 * KCH + kk, KCH); // rows 16..31
      v16bf B0 = load_frag_bf16(b0 + k0 + kk, SHID);
      v16bf B1 = load_frag_bf16(b1 + k0 + kk, SHID);
      c00 = WMMA_BF16(A0, B0, c00); c01 = WMMA_BF16(A0, B1, c01);
      c10 = WMMA_BF16(A1, B0, c10); c11 = WMMA_BF16(A1, B1, c11);
    }
    if (wave == 0) tdm_wait0();
    __syncthreads();
  }
#else
  const __bf16* a0 = asrc;
  const __bf16* a1 = a0 + 16 * SHID;
  for (int k0 = 0; k0 < SHID; k0 += 32) {
    v16bf A0 = load_frag_bf16(a0 + k0, SHID);
    v16bf A1 = load_frag_bf16(a1 + k0, SHID);
    v16bf B0 = load_frag_bf16(b0 + k0, SHID);
    v16bf B1 = load_frag_bf16(b1 + k0, SHID);
    c00 = WMMA_BF16(A0, B0, c00); c01 = WMMA_BF16(A0, B1, c01);
    c10 = WMMA_BF16(A1, B0, c10); c11 = WMMA_BF16(A1, B1, c11);
  }
#endif

  int lane = threadIdx.x & 31, n = lane & 15, half = lane >> 4;
  v8f* C[4] = { &c00, &c01, &c10, &c11 };
#pragma unroll
  for (int q = 0; q < 4; ++q) {
    int r = (q >> 1) * 16, c = (q & 1) * 16;
#pragma unroll
    for (int v = 0; v < 8; ++v) {
      int m = row0 + r + half * 8 + v;
      out[(size_t)m * DIM + col0 + c + n] = (*C[q])[v];
    }
  }
}

// ---------------- routed down (sum over experts) adds into y ----------------
__global__ void routed_down(const __bf16* __restrict__ Hr,   // [E, N, H]
                            const __bf16* __restrict__ wdt,  // [E, D, H]
                            float* __restrict__ out) {       // [N, D]
  int wid = blockIdx.x * (blockDim.x >> 5) + (threadIdx.x >> 5);
  const int cB = DIM / 32;
  int rb = wid / cB, cb = wid % cB;
  if (rb >= N_TOK / 32) return;
  int row0 = rb * 32, col0 = cb * 32;

  v8f c00 = {}, c01 = {}, c10 = {}, c11 = {};
  for (int e = 0; e < NEXP; ++e) {
    const __bf16* a0 = Hr + ((size_t)e * N_TOK + row0) * HID;
    const __bf16* a1 = a0 + 16 * HID;
    const __bf16* b0 = wdt + ((size_t)e * DIM + col0) * HID;
    const __bf16* b1 = b0 + 16 * HID;
    for (int k0 = 0; k0 < HID; k0 += 32) {
      __builtin_prefetch(b0 + k0 + 64, 0, 3);
      v16bf A0 = load_frag_bf16(a0 + k0, HID);
      v16bf A1 = load_frag_bf16(a1 + k0, HID);
      v16bf B0 = load_frag_bf16(b0 + k0, HID);
      v16bf B1 = load_frag_bf16(b1 + k0, HID);
      c00 = WMMA_BF16(A0, B0, c00); c01 = WMMA_BF16(A0, B1, c01);
      c10 = WMMA_BF16(A1, B0, c10); c11 = WMMA_BF16(A1, B1, c11);
    }
  }
  int lane = threadIdx.x & 31, n = lane & 15, half = lane >> 4;
  v8f* C[4] = { &c00, &c01, &c10, &c11 };
#pragma unroll
  for (int q = 0; q < 4; ++q) {
    int r = (q >> 1) * 16, c = (q & 1) * 16;
#pragma unroll
    for (int v = 0; v < 8; ++v) {
      int m = row0 + r + half * 8 + v;
      out[(size_t)m * DIM + col0 + c + n] += (*C[q])[v];
    }
  }
}

// ---------------- aux loss finalize ----------------
__global__ void finalize_aux(const float* __restrict__ aux, float* __restrict__ out_aux) {
  if (threadIdx.x != 0 || blockIdx.x != 0) return;
  float csum = 0.f;
  for (int e = 0; e < NEXP; ++e) csum += aux[16 + e];
  float inv_c = 1.f / fmaxf(csum, 1e-6f);
  float dot = 0.f;
  for (int e = 0; e < NEXP; ++e) {
    float P = aux[e] / (float)N_TOK;
    float f = aux[16 + e] * inv_c;
    dot += f * P;
  }
  float Lbal = (float)NEXP * dot;
  float null_rate = aux[32] / (float)(N_TOK * TOPK);
  float Lnull = (null_rate - RHO) * (null_rate - RHO);
  float Lz = aux[33] / (float)N_TOK;
  out_aux[0] = 0.02f * Lbal + 0.001f * Lz + 0.01f * Lnull;
}

// ---------------- launch ----------------
extern "C" void kernel_launch(void* const* d_in, const int* in_sizes, int n_in,
                              void* d_out, int out_size, void* d_ws, size_t ws_size,
                              hipStream_t stream) {
  const float* x          = (const float*)d_in[0];
  const float* gate_w     = (const float*)d_in[1];
  const float* logit_bias = (const float*)d_in[2];
  const float* null_logit = (const float*)d_in[3];
  const float* sgw        = (const float*)d_in[4];
  const float* suw        = (const float*)d_in[5];
  const float* sdw        = (const float*)d_in[6];
  const float* Wg         = (const float*)d_in[7];
  const float* Wu         = (const float*)d_in[8];
  const float* Wd         = (const float*)d_in[9];
  float* out = (float*)d_out;  // [N*D] y, then [1] aux_loss

  // ---- workspace layout (16B aligned regions) ----
  char* w = (char*)d_ws;
  float* combine = (float*)w;  w += (size_t)N_TOK * NEXP * 4;
  float* aux     = (float*)w;  w += 64 * 4;
  __bf16* xb     = (__bf16*)w; w += (size_t)N_TOK * DIM * 2;
  __bf16* sgwb   = (__bf16*)w; w += (size_t)SHID * DIM * 2;
  __bf16* suwb   = (__bf16*)w; w += (size_t)SHID * DIM * 2;
  __bf16* sdwb   = (__bf16*)w; w += (size_t)DIM * SHID * 2;
  __bf16* wgt    = (__bf16*)w; w += (size_t)NEXP * HID * DIM * 2;
  __bf16* wut    = (__bf16*)w; w += (size_t)NEXP * HID * DIM * 2;
  __bf16* wdt    = (__bf16*)w; w += (size_t)NEXP * DIM * HID * 2;
  __bf16* Hs     = (__bf16*)w; w += (size_t)N_TOK * SHID * 2;
  __bf16* Hr     = (__bf16*)w;

  // ---- one-time bf16 conversions ----
  {
    int n4 = (N_TOK * DIM) / 4;
    hipLaunchKernelGGL(cvt_bf16_v4, dim3((n4 + 255) / 256), dim3(256), 0, stream, x, xb, n4);
  }
  {
    int n4 = (SHID * DIM) / 4;
    hipLaunchKernelGGL(cvt_bf16_v4, dim3((n4 + 255) / 256), dim3(256), 0, stream, sgw, sgwb, n4);
    hipLaunchKernelGGL(cvt_bf16_v4, dim3((n4 + 255) / 256), dim3(256), 0, stream, suw, suwb, n4);
    hipLaunchKernelGGL(cvt_bf16_v4, dim3((n4 + 255) / 256), dim3(256), 0, stream, sdw, sdwb, n4);
  }
  // W_gate/W_up: [E, D, H] -> [E, H, D] ; W_down: [E, H, D] -> [E, D, H]
  hipLaunchKernelGGL(transpose_cvt, dim3(HID / 16, DIM / 16, NEXP), dim3(16, 16), 0, stream,
                     Wg, wgt, DIM, HID);
  hipLaunchKernelGGL(transpose_cvt, dim3(HID / 16, DIM / 16, NEXP), dim3(16, 16), 0, stream,
                     Wu, wut, DIM, HID);
  hipLaunchKernelGGL(transpose_cvt, dim3(DIM / 16, HID / 16, NEXP), dim3(16, 16), 0, stream,
                     Wd, wdt, HID, DIM);

  // ---- router ----
  hipLaunchKernelGGL(init_aux, dim3(1), dim3(64), 0, stream, aux);
  hipLaunchKernelGGL(moe_router, dim3(N_TOK / 8), dim3(256), 0, stream,
                     x, gate_w, logit_bias, null_logit, combine, aux);

  // ---- GEMMs ----
  {
    int waves = (N_TOK / 32) * (SHID / 32);
    hipLaunchKernelGGL(shared_mlp1, dim3(waves / 8), dim3(256), 0, stream, xb, sgwb, suwb, Hs);
  }
  {
    int waves = NEXP * (N_TOK / 32) * (HID / 32);
    hipLaunchKernelGGL(routed_mlp1, dim3(waves / 8), dim3(256), 0, stream, xb, wgt, wut, combine, Hr);
  }
  hipLaunchKernelGGL(shared_down, dim3(N_TOK / 32, (DIM / 32) / 8), dim3(256), 0, stream,
                     Hs, sdwb, out);
  {
    int waves = (N_TOK / 32) * (DIM / 32);
    hipLaunchKernelGGL(routed_down, dim3(waves / 8), dim3(256), 0, stream, Hr, wdt, out);
  }
  hipLaunchKernelGGL(finalize_aux, dim3(1), dim3(32), 0, stream, aux, out + (size_t)N_TOK * DIM);
}
